// GitEncoder_5102421147654
// MI455X (gfx1250) — compile-verified, matrix operand order checked by reading, not run on hardware
//
#include <hip/hip_runtime.h>
#include <hip/hip_bf16.h>
#include <math.h>

// ---------------------------------------------------------------------------
// Types for CDNA5 WMMA (wave32): V_WMMA_F32_16X16X32_BF16
// ---------------------------------------------------------------------------
typedef __attribute__((ext_vector_type(16))) __bf16 v16bf;
typedef __attribute__((ext_vector_type(8)))  __bf16 v8bf;
typedef __attribute__((ext_vector_type(8)))  float  v8f;

__device__ __forceinline__ v8f wmma_bf16(v16bf a, v16bf b, v8f c) {
    // (neg_a, A, neg_b, B, c_mod, C, reuse_a, reuse_b)
    return __builtin_amdgcn_wmma_f32_16x16x32_bf16(false, a, false, b,
                                                   (short)0, c, false, false);
}

// hardware f32 -> bf16 (RNE) via native convert
__device__ __forceinline__ __bf16 f2bf(float f) { return (__bf16)f; }

// LDS byte offset of a __shared__ object: flat LDS addresses carry the DS
// offset in the low 32 bits.
__device__ __forceinline__ unsigned lds_off(const void* p) {
    return (unsigned)(uintptr_t)p;
}

// Wait for this wave's outstanding async (global<->LDS) ops.
__device__ __forceinline__ void wait_async0() {
#if __has_builtin(__builtin_amdgcn_s_wait_asynccnt)
    __builtin_amdgcn_s_wait_asynccnt(0);
#else
    asm volatile("s_wait_asynccnt 0x0" ::: "memory");
#endif
}

// Async copy of 16 bytes global -> LDS (ASYNCcnt-tracked, VGLOBAL GVS mode):
// LDS[ldsByteOff] = MEM[saddr + voffByte], 16B, per active lane.
__device__ __forceinline__ void async_g2l_b128(unsigned ldsByteOff,
                                               unsigned voffByte,
                                               const void* saddr) {
    asm volatile("global_load_async_to_lds_b128 %0, %1, %2"
                 :: "v"(ldsByteOff), "v"(voffByte),
                    "s"((unsigned long long)(uintptr_t)saddr)
                 : "memory");
}

// Load one 16x32 bf16 A/B fragment from LDS (row-major, K contiguous).
// ISA 16-bit A layout: lanes 0-15 hold row m=lane, K {0..7,16..23};
// lanes 16-31 hold row m=lane-16, K {8..15,24..31}. Two 16B ds loads.
__device__ __forceinline__ v16bf ldfrag(const __bf16* p, int stride) {
    const int lane = (int)(threadIdx.x & 31);
    const __bf16* row = p + (lane & 15) * stride + ((lane < 16) ? 0 : 8);
    union { v16bf v; v8bf h[2]; } u;
    u.h[0] = *reinterpret_cast<const v8bf*>(row);
    u.h[1] = *reinterpret_cast<const v8bf*>(row + 16);
    return u.v;
}

// ---------------------------------------------------------------------------
// Weight convert + transpose: W fp32 [K,N] -> Wt bf16 [N,K].
// 32x32 tiles through LDS; reads coalesced along N, writes coalesced along K.
// ---------------------------------------------------------------------------
__global__ __launch_bounds__(256)
void wtrans_kernel(const float* __restrict__ W, __bf16* __restrict__ Wt,
                   int K, int N) {
    const int n0 = (int)blockIdx.x * 32, k0 = (int)blockIdx.y * 32;
    const int tid = (int)threadIdx.x;
    __shared__ __bf16 t[32][40];
#pragma unroll
    for (int it = 0; it < 4; ++it) {
        int idx = it * 256 + tid;
        int kk = idx >> 5, nn = idx & 31;
        t[nn][kk] = f2bf(W[(k0 + kk) * N + n0 + nn]);
    }
    __syncthreads();
#pragma unroll
    for (int it = 0; it < 4; ++it) {
        int idx = it * 256 + tid;
        int nn = idx >> 5, kk = idx & 31;
        Wt[(n0 + nn) * K + k0 + kk] = t[nn][kk];
    }
}

// ---------------------------------------------------------------------------
// bf16-WMMA GEMM: out[M,N] = epilogue(A[M,K] @ Wt^T + bias [+res])
//   A  : fp32 [M,K], converted to bf16 while staging to LDS
//   Wt : bf16 [N,K] (pre-transposed) -> B tile staged raw with
//        GLOBAL_LOAD_ASYNC_TO_LDS_B128 (ASYNCcnt), no conversion in-loop.
// Block tile 128x128, 256 threads = 8 waves; wave tile 32x64 (2x4 WMMA tiles).
// M % 128 == 0, N % 128 == 0, K % 32 == 0.
// ---------------------------------------------------------------------------
template <bool HAS_RES, bool GELU>
__global__ __launch_bounds__(256)
void gemm_bf16_kernel(const float* __restrict__ A, const __bf16* __restrict__ Wt,
                      const float* __restrict__ bias, const float* __restrict__ res,
                      float* __restrict__ out, int M, int N, int K) {
    const int bm = blockIdx.y, bn = blockIdx.x;
    const int tid = (int)threadIdx.x;
    const int wid = tid >> 5, lane = tid & 31;
    const int wm = wid & 3;   // 4 waves along M, 32 rows each
    const int wn = wid >> 2;  // 2 waves along N, 64 cols each

    const int AS = 40;        // LDS row stride (bf16 elems): 80B, 16B-aligned
    __shared__ __align__(16) __bf16 lA[128 * 40];  // A tile 128x32
    __shared__ __align__(16) __bf16 lB[128 * 40];  // Wt tile [n][k] 128x32

    const unsigned lB_base = lds_off(&lB[0]);

    v8f acc[2][4];
    const v8f vzero = {0.f, 0.f, 0.f, 0.f, 0.f, 0.f, 0.f, 0.f};
#pragma unroll
    for (int mt = 0; mt < 2; ++mt)
#pragma unroll
        for (int nt = 0; nt < 4; ++nt) acc[mt][nt] = vzero;

    for (int k0 = 0; k0 < K; k0 += 32) {
        // --- stage B: async raw copy of 128 rows x 32 k bf16 (512 x 16B)
#pragma unroll
        for (int it = 0; it < 2; ++it) {
            int seg = it * 256 + tid;      // 4 segments of 8 bf16 per row
            int n = seg >> 2;
            int k8 = (seg & 3) << 3;
            unsigned voff = (unsigned)(((bn * 128 + n) * K + k0 + k8) * 2);
            unsigned loff = lB_base + (unsigned)((n * AS + k8) * 2);
            async_g2l_b128(loff, voff, Wt);
        }
        // --- stage A: 128 rows x 32 k  (1024 float4, 4 per thread), cvt bf16
#pragma unroll
        for (int it = 0; it < 4; ++it) {
            int idx = it * 256 + tid;
            int r = idx >> 3;
            int c4 = (idx & 7) << 2;
            const float4 f = *reinterpret_cast<const float4*>(
                &A[(bm * 128 + r) * K + k0 + c4]);
            __bf16* d = &lA[r * AS + c4];
            d[0] = f2bf(f.x); d[1] = f2bf(f.y); d[2] = f2bf(f.z); d[3] = f2bf(f.w);
        }
        wait_async0();
        __syncthreads();

        v16bf aF[2], bF[4];
#pragma unroll
        for (int mt = 0; mt < 2; ++mt)
            aF[mt] = ldfrag(&lA[(wm * 32 + mt * 16) * AS], AS);
#pragma unroll
        for (int nt = 0; nt < 4; ++nt)
            bF[nt] = ldfrag(&lB[(wn * 64 + nt * 16) * AS], AS);
#pragma unroll
        for (int mt = 0; mt < 2; ++mt)
#pragma unroll
            for (int nt = 0; nt < 4; ++nt)
                acc[mt][nt] = wmma_bf16(aF[mt], bF[nt], acc[mt][nt]);
        __syncthreads();
    }

    // Epilogue. C layout: lanes 0-15 cols 0-15 rows mb+0..7 (mb=0);
    // lanes 16-31 same cols, rows 8..15 (mb=8). 32-bit index math only.
    const int col16 = lane & 15;
    const int mb = (lane < 16) ? 0 : 8;
#pragma unroll
    for (int mt = 0; mt < 2; ++mt) {
        const int row0 = bm * 128 + wm * 32 + mt * 16 + mb;
#pragma unroll
        for (int nt = 0; nt < 4; ++nt) {
            const int gcol = bn * 128 + wn * 64 + nt * 16 + col16;
            const float bsv = bias[gcol];
            int off = row0 * N + gcol;
#pragma unroll
            for (int i = 0; i < 8; ++i, off += N) {
                float v = acc[mt][nt][i] + bsv;
                if (HAS_RES) v += res[off];
                if (GELU) v = 0.5f * v * (1.0f + erff(v * 0.70710678118654752f));
                out[off] = v;
            }
        }
    }
}

// ---------------------------------------------------------------------------
// Flash attention with WMMA (online softmax, no SxS materialization).
// One block = (b, h, 64 q-rows); 4 waves x 16 q-rows. Key chunks of 32.
// Q/K/V fp32 [B*S, D] with head slice at col h*64.
// ---------------------------------------------------------------------------
__global__ __launch_bounds__(128)
void attn_flash_kernel(const float* __restrict__ Q, const float* __restrict__ Kg,
                       const float* __restrict__ Vg, const float* __restrict__ mask,
                       float* __restrict__ ctx) {
    const int S = 1024, D = 768, DH = 64;
    const int bx = (int)blockIdx.x;
    const int qblk = bx & 15;
    const int h = (bx >> 4) % 12;
    const int b = bx / (16 * 12);
    const int tid = (int)threadIdx.x, wid = tid >> 5, lane = tid & 31;
    const int qbase = qblk * 64 + wid * 16;

    const int KT_S = 72;  // lK row stride (bf16): 144B, 16B multiple
    const int VT_S = 40;  // lV / lS row stride: 80B
    __shared__ __align__(16) __bf16 lK[32 * 72];      // [key][d]   32x64
    __shared__ __align__(16) __bf16 lV[64 * 40];      // [d][key]   64x32 (transposed)
    __shared__ __align__(16) __bf16 lS[4][16 * 40];   // per-wave P tile 16x32

    // Load Q tile into two A fragments (d=0..31, d=32..63), fp32 -> bf16.
    const int r = lane & 15;
    const int kb = (lane < 16) ? 0 : 8;
    const float* qrow = Q + (b * S + qbase + r) * D + h * DH;
    union uf { v16bf v; __bf16 s[16]; };
    uf qa[2];
#pragma unroll
    for (int j = 0; j < 2; ++j)
#pragma unroll
        for (int e = 0; e < 16; ++e) {
            const int d = j * 32 + kb + ((e < 8) ? e : 16 + (e - 8));
            qa[j].s[e] = f2bf(qrow[d]);
        }

    float mrow[8], lrow[8];
    v8f o[4];
    const v8f vzero = {0.f, 0.f, 0.f, 0.f, 0.f, 0.f, 0.f, 0.f};
#pragma unroll
    for (int i = 0; i < 8; ++i) { mrow[i] = -3.0e38f; lrow[i] = 0.f; }
#pragma unroll
    for (int nt = 0; nt < 4; ++nt) o[nt] = vzero;

    const int col16 = lane & 15;
    const int mb = (lane < 16) ? 0 : 8;

    for (int kv0 = 0; kv0 < S; kv0 += 32) {
        // --- stage K chunk [key][d] and V chunk transposed [d][key]
#pragma unroll
        for (int it = 0; it < 4; ++it) {
            int idx = it * 128 + tid;
            int key = idx >> 4;
            int d4 = (idx & 15) << 2;
            const int goff = (b * S + kv0 + key) * D + h * DH + d4;
            const float4 kf = *reinterpret_cast<const float4*>(&Kg[goff]);
            __bf16* dk = &lK[key * KT_S + d4];
            dk[0] = f2bf(kf.x); dk[1] = f2bf(kf.y); dk[2] = f2bf(kf.z); dk[3] = f2bf(kf.w);
            const float4 vf = *reinterpret_cast<const float4*>(&Vg[goff]);
            lV[(d4 + 0) * VT_S + key] = f2bf(vf.x);
            lV[(d4 + 1) * VT_S + key] = f2bf(vf.y);
            lV[(d4 + 2) * VT_S + key] = f2bf(vf.z);
            lV[(d4 + 3) * VT_S + key] = f2bf(vf.w);
        }
        __syncthreads();

        // --- scores: S16x32 = Q(16x64) . K^T(64x32), two n-tiles of 16 keys
        v8f sc[2];
#pragma unroll
        for (int nt = 0; nt < 2; ++nt) {
            v16bf bk0 = ldfrag(&lK[(nt * 16) * KT_S], KT_S);       // d 0..31
            v16bf bk1 = ldfrag(&lK[(nt * 16) * KT_S + 32], KT_S);  // d 32..63
            v8f t = wmma_bf16(qa[0].v, bk0, vzero);
            sc[nt] = wmma_bf16(qa[1].v, bk1, t);
        }

        // --- online softmax update
        const float mk0 = mask[b * S + kv0 + col16];
        const float mk1 = mask[b * S + kv0 + 16 + col16];
        float sv[2][8], rm[8];
#pragma unroll
        for (int i = 0; i < 8; ++i) {
            sv[0][i] = sc[0][i] * 0.125f + mk0;
            sv[1][i] = sc[1][i] * 0.125f + mk1;
            rm[i] = fmaxf(sv[0][i], sv[1][i]);
        }
#pragma unroll
        for (int off = 1; off < 16; off <<= 1)
#pragma unroll
            for (int i = 0; i < 8; ++i)
                rm[i] = fmaxf(rm[i], __shfl_xor(rm[i], off, 16));

        float corr[8], ls[8];
#pragma unroll
        for (int i = 0; i < 8; ++i) {
            const float mn = fmaxf(mrow[i], rm[i]);
            corr[i] = __expf(mrow[i] - mn);
            mrow[i] = mn;
            const float p0 = __expf(sv[0][i] - mn);
            const float p1 = __expf(sv[1][i] - mn);
            lS[wid][(mb + i) * VT_S + col16] = f2bf(p0);
            lS[wid][(mb + i) * VT_S + 16 + col16] = f2bf(p1);
            ls[i] = p0 + p1;
        }
#pragma unroll
        for (int off = 1; off < 16; off <<= 1)
#pragma unroll
            for (int i = 0; i < 8; ++i)
                ls[i] += __shfl_xor(ls[i], off, 16);
#pragma unroll
        for (int i = 0; i < 8; ++i) lrow[i] = lrow[i] * corr[i] + ls[i];
#pragma unroll
        for (int nt = 0; nt < 4; ++nt)
#pragma unroll
            for (int i = 0; i < 8; ++i) o[nt][i] *= corr[i];

        // --- O += P(16x32) . V(32x64), four n-tiles of 16 d-cols
        v16bf aP = ldfrag(&lS[wid][0], VT_S);
#pragma unroll
        for (int nt = 0; nt < 4; ++nt) {
            v16bf bv = ldfrag(&lV[(nt * 16) * VT_S], VT_S);
            o[nt] = wmma_bf16(aP, bv, o[nt]);
        }
        __syncthreads();
    }

    // write ctx, normalized by row sums
#pragma unroll
    for (int nt = 0; nt < 4; ++nt)
#pragma unroll
        for (int i = 0; i < 8; ++i) {
            const int row = qbase + mb + i;
            ctx[(b * S + row) * D + h * DH + nt * 16 + col16] = o[nt][i] / lrow[i];
        }
}

// ---------------------------------------------------------------------------
// LayerNorm over D=768, one block per row. EPS = 1e-12.
// ---------------------------------------------------------------------------
__global__ __launch_bounds__(256)
void ln_kernel(const float* __restrict__ x, const float* __restrict__ g,
               const float* __restrict__ bta, float* __restrict__ out) {
    const int row = (int)blockIdx.x, tid = (int)threadIdx.x;
    __shared__ float red[256];
    __shared__ float stat;
    const float* xr = x + row * 768;
    const float v0 = xr[tid], v1 = xr[tid + 256], v2 = xr[tid + 512];
    red[tid] = v0 + v1 + v2;
    __syncthreads();
    for (int o = 128; o > 0; o >>= 1) {
        if (tid < o) red[tid] += red[tid + o];
        __syncthreads();
    }
    if (tid == 0) stat = red[0] * (1.0f / 768.0f);
    __syncthreads();
    const float mean = stat;
    const float d0 = v0 - mean, d1 = v1 - mean, d2 = v2 - mean;
    __syncthreads();
    red[tid] = d0 * d0 + d1 * d1 + d2 * d2;
    __syncthreads();
    for (int o = 128; o > 0; o >>= 1) {
        if (tid < o) red[tid] += red[tid + o];
        __syncthreads();
    }
    if (tid == 0) stat = red[0] * (1.0f / 768.0f);
    __syncthreads();
    const float inv = rsqrtf(stat + 1e-12f);
    float* orow = out + row * 768;
    orow[tid]       = d0 * inv * g[tid]       + bta[tid];
    orow[tid + 256] = d1 * inv * g[tid + 256] + bta[tid + 256];
    orow[tid + 512] = d2 * inv * g[tid + 512] + bta[tid + 512];
}

// ---------------------------------------------------------------------------
// Driver: 6 layers, all on `stream`.
// Workspace layout, MD = 4096*768 floats:
//   fp32: h@0 | q@1MD | k@2MD (->attn_out) | v@3MD (inter spans 3MD..7MD) | ctx@4MD
//   bf16 (after 7MD floats): per-layer transposed weights, reused each layer:
//     wtq|wtk|wtv|wto (D*D each) | wti (F*D) | wtf (D*F)  -> 7,077,888 bf16
// Required ws: 7*MD*4 + 14.2MB = ~102.3 MB.
// ---------------------------------------------------------------------------
extern "C" void kernel_launch(void* const* d_in, const int* in_sizes, int n_in,
                              void* d_out, int out_size, void* d_ws, size_t ws_size,
                              hipStream_t stream) {
    (void)in_sizes; (void)n_in; (void)out_size; (void)ws_size;
    const int L = 6, D = 768, F = 3072, Mrows = 4096;

    const float* h_in = (const float*)d_in[0];
    const float* mask = (const float*)d_in[1];
    const float* Wq = (const float*)d_in[2];
    const float* bq = (const float*)d_in[3];
    const float* Wk = (const float*)d_in[4];
    const float* bk = (const float*)d_in[5];
    const float* Wv = (const float*)d_in[6];
    const float* bv = (const float*)d_in[7];
    const float* Wo = (const float*)d_in[8];
    const float* bo = (const float*)d_in[9];
    const float* g1 = (const float*)d_in[10];
    const float* b1 = (const float*)d_in[11];
    const float* Wi = (const float*)d_in[12];
    const float* bi = (const float*)d_in[13];
    const float* Wf = (const float*)d_in[14];
    const float* bf_ = (const float*)d_in[15];
    const float* g2 = (const float*)d_in[16];
    const float* b2 = (const float*)d_in[17];

    float* ws = (float*)d_ws;
    const size_t MD = (size_t)Mrows * D;
    float* Hb  = ws;            // hidden state between layers
    float* Qb  = ws + 1 * MD;
    float* Kb  = ws + 2 * MD;
    float* Vb  = ws + 3 * MD;
    float* Cb  = ws + 4 * MD;   // attention context
    float* Tb  = Qb;            // residual-sum temp (q dead by then)
    float* AOb = Kb;            // attn_out after LN1 (k dead by then)
    float* Ib  = Vb;            // intermediate [M,F], spans 3MD..7MD (v,ctx dead)

    __bf16* wt  = (__bf16*)(ws + 7 * MD);  // 16B-aligned (7*MD*4 % 16 == 0)
    __bf16* wtq = wt;
    __bf16* wtk = wtq + (size_t)D * D;
    __bf16* wtv = wtk + (size_t)D * D;
    __bf16* wto = wtv + (size_t)D * D;
    __bf16* wti = wto + (size_t)D * D;     // [F, D]
    __bf16* wtf = wti + (size_t)F * D;     // [D, F]

    const dim3 blk(256);
    const dim3 gD(D / 128, Mrows / 128);   // N=768  -> 6 x 32
    const dim3 gF(F / 128, Mrows / 128);   // N=3072 -> 24 x 32
    const dim3 tDD(D / 32, D / 32);        // 24 x 24
    const dim3 tDF(F / 32, D / 32);        // W [D,F] -> Wt [F,D]: 96 x 24
    const dim3 tFD(D / 32, F / 32);        // W [F,D] -> Wt [D,F]: 24 x 96

    for (int l = 0; l < L; ++l) {
        const float* h = (l == 0) ? h_in : Hb;
        const float* wq = Wq + (size_t)l * D * D;
        const float* wk = Wk + (size_t)l * D * D;
        const float* wv = Wv + (size_t)l * D * D;
        const float* wo = Wo + (size_t)l * D * D;
        const float* wi = Wi + (size_t)l * D * F;
        const float* wf = Wf + (size_t)l * F * D;

        // per-layer weight convert+transpose to bf16 [N][K]
        wtrans_kernel<<<tDD, blk, 0, stream>>>(wq, wtq, D, D);
        wtrans_kernel<<<tDD, blk, 0, stream>>>(wk, wtk, D, D);
        wtrans_kernel<<<tDD, blk, 0, stream>>>(wv, wtv, D, D);
        wtrans_kernel<<<tDD, blk, 0, stream>>>(wo, wto, D, D);
        wtrans_kernel<<<tDF, blk, 0, stream>>>(wi, wti, D, F);
        wtrans_kernel<<<tFD, blk, 0, stream>>>(wf, wtf, F, D);

        // QKV projections (no residual, no act)
        gemm_bf16_kernel<false, false><<<gD, blk, 0, stream>>>(
            h, wtq, bq + l * D, nullptr, Qb, Mrows, D, D);
        gemm_bf16_kernel<false, false><<<gD, blk, 0, stream>>>(
            h, wtk, bk + l * D, nullptr, Kb, Mrows, D, D);
        gemm_bf16_kernel<false, false><<<gD, blk, 0, stream>>>(
            h, wtv, bv + l * D, nullptr, Vb, Mrows, D, D);
        // attention: B*H*(S/64) = 4*12*16 = 768 blocks
        attn_flash_kernel<<<768, 128, 0, stream>>>(Qb, Kb, Vb, mask, Cb);
        // self-output dense + residual
        gemm_bf16_kernel<true, false><<<gD, blk, 0, stream>>>(
            Cb, wto, bo + l * D, h, Tb, Mrows, D, D);
        ln_kernel<<<Mrows, blk, 0, stream>>>(Tb, g1 + l * D, b1 + l * D, AOb);
        // FFN: dense+GELU, dense+residual
        gemm_bf16_kernel<false, true><<<gF, blk, 0, stream>>>(
            AOb, wti, bi + l * F, nullptr, Ib, Mrows, F, D);
        gemm_bf16_kernel<true, false><<<gD, blk, 0, stream>>>(
            Ib, wtf, bf_ + l * D, AOb, Tb, Mrows, D, F);
        float* dst = (l == L - 1) ? (float*)d_out : Hb;
        ln_kernel<<<Mrows, blk, 0, stream>>>(Tb, g2 + l * D, b2 + l * D, dst);
    }
}
